// DualMambaStream_64785286693353
// MI455X (gfx1250) — compile-verified, matrix operand order checked by reading, not run on hardware
//
#include <hip/hip_runtime.h>
#include <hip/hip_bf16.h>

// ---------------------------------------------------------------------------
// Dual Mamba block for MI455X (gfx1250): bf16 WMMA GEMMs with async global->LDS
// double-buffered staging + VALU selective scan.
// ---------------------------------------------------------------------------

typedef __bf16 bf16;
typedef __attribute__((ext_vector_type(16))) __bf16 bf16x16;
typedef __attribute__((ext_vector_type(8)))  float  v8f;

// GCC-style vector matching the builtin's expected pointee type
// ('__attribute__((__vector_size__(4 * sizeof(int)))) int' per the clang diag).
typedef int v4i_vs __attribute__((vector_size(16)));

union frag_u {
    uint4   q[2];   // 32 bytes
    bf16x16 v;      // 16 x bf16 = 32 bytes
};
static_assert(sizeof(frag_u) == 32, "frag size");

// Problem constants
constexpr int L_SEQ = 2048;
constexpr int DMODEL = 1024;
constexpr int DI_ID = 2048, NST_ID = 16, R_ID = 96;   // dt_rank(64) + 2*16
constexpr int DI_AT = 1024, NST_AT = 8,  R_AT = 80;   // dt_rank(64) + 2*8
constexpr int DTR = 64;

// ---------------------------------------------------------------------------
// gfx1250 async global->LDS copy (ASYNCcnt path), with safe fallback.
// ---------------------------------------------------------------------------
#if __has_builtin(__builtin_amdgcn_global_load_async_to_lds_b128)
#define USE_ASYNC_LDS 1
#else
#define USE_ASYNC_LDS 0
#endif

__device__ __forceinline__ void cp16(bf16* lds_dst, const bf16* gsrc)
{
#if USE_ASYNC_LDS
    __builtin_amdgcn_global_load_async_to_lds_b128(
        (__attribute__((address_space(1))) v4i_vs*)gsrc,
        (__attribute__((address_space(3))) v4i_vs*)lds_dst, 0, 0);
#else
    *(uint4*)lds_dst = *(const uint4*)gsrc;
#endif
}

__device__ __forceinline__ void wait_async_0()
{
#if __has_builtin(__builtin_amdgcn_s_wait_asynccnt)
    __builtin_amdgcn_s_wait_asynccnt(0);
#elif defined(__AMDGCN__)
    asm volatile("s_wait_asynccnt 0x0" ::: "memory");
#endif
}

__device__ __forceinline__ void wait_async_4()
{
#if __has_builtin(__builtin_amdgcn_s_wait_asynccnt)
    __builtin_amdgcn_s_wait_asynccnt(4);
#elif defined(__AMDGCN__)
    asm volatile("s_wait_asynccnt 0x4" ::: "memory");
#endif
}

// ---------------------------------------------------------------------------
// GEMM: C[M,N] = A[M,K] (bf16, row-major) * W[N,K]^T (bf16, row-major)
// Block tile 128x128, 8 waves (4M x 2N); each wave computes 32x64 via a
// 2x4 grid of v_wmma_f32_16x16x32_bf16. K staged in 32-wide slabs through
// double-buffered LDS, filled with async global->LDS copies that overlap WMMA.
// M % 128 == 0 and K % 32 == 0 hold for all call sites; N is guarded.
// ---------------------------------------------------------------------------
#define LDS_PAD 48   // 32 elements + 16 pad (96 B rows, 16B-aligned, bank-spread)

__global__ __launch_bounds__(256) void gemm_bf16_wmma(
    const bf16* __restrict__ A, const bf16* __restrict__ Bw,
    float* __restrict__ C, int M, int N, int K)
{
    __shared__ __align__(16) bf16 sA[2][128][LDS_PAD];   // 2 x 12 KB
    __shared__ __align__(16) bf16 sB[2][128][LDS_PAD];   // 2 x 12 KB

    const int tid   = threadIdx.x;
    const int lane  = tid & 31;
    const int wave  = tid >> 5;
    const int waveM = wave >> 1;     // 0..3  -> rows of 32
    const int waveN = wave & 1;      // 0..1  -> cols of 64
    const int tileM0 = blockIdx.y * 128;
    const int tileN0 = blockIdx.x * 128;

    // 16-bit A/B fragment addressing: lanes 0-15 -> rows, lane[4] selects
    // which K-halves (k = fk..fk+7 and fk+16..fk+23) the lane holds.
    const int frow = lane & 15;
    const int fk   = (lane >> 4) * 8;

    v8f acc[2][4] = {};

    // Stage one 128x32 slab of A and of W into LDS buffer `buf`.
    // 512 16B segments per matrix; 256 threads x 2 each => 4 copies/thread.
    auto stage = [&](int k0, int buf) {
        #pragma unroll
        for (int i = 0; i < 2; ++i) {
            int s  = tid + i * 256;
            int r  = s >> 2;
            int cs = (s & 3) * 8;
            cp16(&sA[buf][r][cs], A + (size_t)(tileM0 + r) * K + (k0 + cs));
        }
        #pragma unroll
        for (int i = 0; i < 2; ++i) {
            int s  = tid + i * 256;
            int r  = s >> 2;
            int cs = (s & 3) * 8;
            int gn = tileN0 + r;
            if (gn < N) {
                cp16(&sB[buf][r][cs], Bw + (size_t)gn * K + (k0 + cs));
            } else {
                uint4 z = make_uint4(0u, 0u, 0u, 0u);
                *(uint4*)&sB[buf][r][cs] = z;   // zero-fill OOB rows
            }
        }
    };

    const int nk = K >> 5;   // K / 32 slabs
    stage(0, 0);

    for (int kk = 0; kk < nk; ++kk) {
        const int cur = kk & 1;
        const bool hasNext = (kk + 1) < nk;
        if (hasNext) {
            stage((kk + 1) << 5, cur ^ 1);
            // prefetch the slab after next into the cache hierarchy
            if (kk + 2 < nk)
                __builtin_prefetch(
                    A + (size_t)(tileM0 + (tid >> 1)) * K + ((kk + 2) << 5), 0, 1);
            wait_async_4();   // our 4 copies for slab kk are done (in-order)
        } else {
            wait_async_0();
        }
        __syncthreads();      // all waves' copies for slab kk visible

        frag_u aF[2], bF[4];
        #pragma unroll
        for (int i = 0; i < 2; ++i) {
            int r = waveM * 32 + i * 16 + frow;
            aF[i].q[0] = *(const uint4*)&sA[cur][r][fk];
            aF[i].q[1] = *(const uint4*)&sA[cur][r][fk + 16];
        }
        #pragma unroll
        for (int j = 0; j < 4; ++j) {
            int r = waveN * 64 + j * 16 + frow;
            bF[j].q[0] = *(const uint4*)&sB[cur][r][fk];
            bF[j].q[1] = *(const uint4*)&sB[cur][r][fk + 16];
        }
        #pragma unroll
        for (int i = 0; i < 2; ++i)
            #pragma unroll
            for (int j = 0; j < 4; ++j)
                acc[i][j] = __builtin_amdgcn_wmma_f32_16x16x32_bf16(
                    false, aF[i].v, false, bF[j].v,
                    (short)0, acc[i][j], false, false);

        __syncthreads();      // everyone finished reading buf[cur]
    }

    // ---- store: C/D layout — lane[3:0]=N, vgpr v + 8*lane[4] = M ----
    const int cn    = lane & 15;
    const int rbase = (lane >> 4) * 8;
    #pragma unroll
    for (int i = 0; i < 2; ++i)
        #pragma unroll
        for (int j = 0; j < 4; ++j) {
            int gm = tileM0 + waveM * 32 + i * 16 + rbase;
            int gn = tileN0 + waveN * 64 + j * 16 + cn;
            if (gn < N) {
                #pragma unroll
                for (int v = 0; v < 8; ++v)
                    C[(size_t)(gm + v) * N + gn] = acc[i][j][v];
            }
        }
}

// ---------------------------------------------------------------------------
// LayerNorm over rows of 1024, output bf16 for the WMMA in-projection.
// ---------------------------------------------------------------------------
__global__ __launch_bounds__(256) void layernorm_to_bf16(
    const float* __restrict__ x, const float* __restrict__ g,
    const float* __restrict__ b, bf16* __restrict__ out)
{
    __shared__ float ssum[256], ssq[256];
    const int row = blockIdx.x;
    const float* xr = x + (size_t)row * DMODEL;
    float s = 0.f, q = 0.f;
    for (int i = threadIdx.x; i < DMODEL; i += 256) {
        float v = xr[i]; s += v; q += v * v;
    }
    ssum[threadIdx.x] = s; ssq[threadIdx.x] = q;
    __syncthreads();
    for (int offd = 128; offd > 0; offd >>= 1) {
        if (threadIdx.x < offd) {
            ssum[threadIdx.x] += ssum[threadIdx.x + offd];
            ssq[threadIdx.x]  += ssq[threadIdx.x + offd];
        }
        __syncthreads();
    }
    float mu  = ssum[0] * (1.0f / DMODEL);
    float var = ssq[0] * (1.0f / DMODEL) - mu * mu;
    float rs  = rsqrtf(var + 1e-5f);
    for (int i = threadIdx.x; i < DMODEL; i += 256)
        out[(size_t)row * DMODEL + i] = (bf16)((xr[i] - mu) * rs * g[i] + b[i]);
}

// Causal depthwise conv (k=4) + bias + SiLU on the x half of xz[L, 2*Di].
__global__ __launch_bounds__(256) void conv_silu(
    const float* __restrict__ xz, const float* __restrict__ cw,
    const float* __restrict__ cb, float* __restrict__ xi_f,
    bf16* __restrict__ xi_b, int Di, int n)
{
    int idx = blockIdx.x * blockDim.x + threadIdx.x;
    if (idx >= n) return;
    int t = idx / Di, c = idx - t * Di;
    float acc = cb[c];
    #pragma unroll
    for (int j = 0; j < 4; ++j) {
        int tt = t - 3 + j;
        if (tt >= 0) acc += cw[c * 4 + j] * xz[(size_t)tt * (2 * Di) + c];
    }
    float sv = acc / (1.f + __expf(-acc));   // SiLU
    xi_f[idx] = sv;
    xi_b[idx] = (bf16)sv;
}

__global__ __launch_bounds__(256) void cvt_f32_bf16(
    const float* __restrict__ in, bf16* __restrict__ out, int n)
{
    int i = blockIdx.x * blockDim.x + threadIdx.x;
    if (i < n) out[i] = (bf16)in[i];
}

// Extract dt columns (first 64 of dbl[L,R]) as bf16 for the dt GEMM.
__global__ __launch_bounds__(256) void cvt_dt_slice(
    const float* __restrict__ dbl, bf16* __restrict__ out, int R)
{
    int i = blockIdx.x * blockDim.x + threadIdx.x;
    if (i >= L_SEQ * DTR) return;
    int t = i >> 6, r = i & 63;
    out[i] = (bf16)dbl[(size_t)t * R + r];
}

__global__ __launch_bounds__(256) void softplus_bias(
    float* __restrict__ dp, const float* __restrict__ b, int Di, int n)
{
    int i = blockIdx.x * blockDim.x + threadIdx.x;
    if (i >= n) return;
    int c = i % Di;
    float v = dp[i] + b[c];
    dp[i] = (v > 20.f) ? v : log1pf(__expf(v));
}

// Selective scan: one thread per channel, sequential over L, nst states.
__global__ __launch_bounds__(256) void scan_kernel(
    const float* __restrict__ delta, const float* __restrict__ xi_f,
    const float* __restrict__ dbl, const float* __restrict__ A_log,
    float* __restrict__ ys, int Di, int nst, int R)
{
    int c = blockIdx.x * blockDim.x + threadIdx.x;
    if (c >= Di) return;
    float Aw[16], h[16];
    for (int j = 0; j < nst; ++j) {
        Aw[j] = -__expf(A_log[(size_t)c * nst + j]);
        h[j]  = 0.f;
    }
    for (int t = 0; t < L_SEQ; ++t) {
        float d  = delta[(size_t)t * Di + c];
        float u  = xi_f[(size_t)t * Di + c];
        float du = d * u;
        const float* bc = dbl + (size_t)t * R + DTR;  // B then C
        float y = 0.f;
        for (int j = 0; j < nst; ++j) {
            h[j] = __expf(d * Aw[j]) * h[j] + du * bc[j];
            y += h[j] * bc[nst + j];
        }
        ys[(size_t)t * Di + c] = y;
    }
}

// y = (ys + xi*D) * silu(z); write bf16 for the output GEMM.
__global__ __launch_bounds__(256) void gate_kernel(
    const float* __restrict__ ys, const float* __restrict__ xi_f,
    const float* __restrict__ Dv, const float* __restrict__ xz,
    bf16* __restrict__ y_b, int Di, int n)
{
    int idx = blockIdx.x * blockDim.x + threadIdx.x;
    if (idx >= n) return;
    int t = idx / Di, c = idx - t * Di;
    float z = xz[(size_t)t * (2 * Di) + Di + c];
    float y = (ys[idx] + xi_f[idx] * Dv[c]) * (z / (1.f + __expf(-z)));
    y_b[idx] = (bf16)y;
}

__global__ __launch_bounds__(256) void residual_kernel(
    const float* __restrict__ seq, const float* __restrict__ outp,
    float* __restrict__ dst, int n)
{
    int i = blockIdx.x * blockDim.x + threadIdx.x;
    if (i < n) dst[i] = seq[i] + 0.1f * outp[i];
}

// ---------------------------------------------------------------------------
// Host side
// ---------------------------------------------------------------------------
static inline dim3 g1(long long n) { return dim3((unsigned)((n + 255) / 256)); }

static void run_stream(hipStream_t stream,
    const float* seq, const float* ln_g, const float* ln_b,
    const float* in_w, const float* conv_w, const float* conv_b,
    const float* xproj_w, const float* dt_w, const float* dt_b,
    const float* A_log, const float* Dv, const float* out_w,
    int Di, int nst, int R,
    bf16* w_in_bf, bf16* w_xproj_bf, bf16* w_dt_bf, bf16* w_out_bf,
    bf16* xn_bf, float* xz, float* xi_f, bf16* xi_b,
    float* dbl, bf16* dt_bf, float* delta, float* ys, bf16* y_bf,
    float* outbuf, float* finaldst)
{
    const int L = L_SEQ, D = DMODEL;
    // weight conversions (f32 -> bf16)
    cvt_f32_bf16<<<g1((long long)2 * Di * D), 256, 0, stream>>>(in_w, w_in_bf, 2 * Di * D);
    cvt_f32_bf16<<<g1((long long)R * Di), 256, 0, stream>>>(xproj_w, w_xproj_bf, R * Di);
    cvt_f32_bf16<<<g1((long long)Di * DTR), 256, 0, stream>>>(dt_w, w_dt_bf, Di * DTR);
    cvt_f32_bf16<<<g1((long long)D * Di), 256, 0, stream>>>(out_w, w_out_bf, D * Di);

    layernorm_to_bf16<<<L, 256, 0, stream>>>(seq, ln_g, ln_b, xn_bf);

    // xz = xn @ in_w^T   [L, 2*Di]
    gemm_bf16_wmma<<<dim3((2 * Di + 127) / 128, L / 128), 256, 0, stream>>>(
        xn_bf, w_in_bf, xz, L, 2 * Di, D);

    conv_silu<<<g1((long long)L * Di), 256, 0, stream>>>(
        xz, conv_w, conv_b, xi_f, xi_b, Di, L * Di);

    // dbl = xi @ xproj_w^T   [L, R]
    gemm_bf16_wmma<<<dim3((R + 127) / 128, L / 128), 256, 0, stream>>>(
        xi_b, w_xproj_bf, dbl, L, R, Di);

    cvt_dt_slice<<<g1((long long)L * DTR), 256, 0, stream>>>(dbl, dt_bf, R);

    // delta_pre = dt @ dt_w^T   [L, Di]
    gemm_bf16_wmma<<<dim3((Di + 127) / 128, L / 128), 256, 0, stream>>>(
        dt_bf, w_dt_bf, delta, L, Di, DTR);

    softplus_bias<<<g1((long long)L * Di), 256, 0, stream>>>(delta, dt_b, Di, L * Di);

    scan_kernel<<<g1(Di), 256, 0, stream>>>(delta, xi_f, dbl, A_log, ys, Di, nst, R);

    gate_kernel<<<g1((long long)L * Di), 256, 0, stream>>>(
        ys, xi_f, Dv, xz, y_bf, Di, L * Di);

    // out = y @ out_w^T   [L, D]
    gemm_bf16_wmma<<<dim3((D + 127) / 128, L / 128), 256, 0, stream>>>(
        y_bf, w_out_bf, outbuf, L, D, Di);

    residual_kernel<<<g1((long long)L * D), 256, 0, stream>>>(
        seq, outbuf, finaldst, L * D);
}

extern "C" void kernel_launch(void* const* d_in, const int* in_sizes, int n_in,
                              void* d_out, int out_size, void* d_ws, size_t ws_size,
                              hipStream_t stream)
{
    (void)in_sizes; (void)n_in; (void)out_size; (void)ws_size;
    const float* id_seq    = (const float*)d_in[0];
    const float* at_seq    = (const float*)d_in[1];
    const float* id_in_w   = (const float*)d_in[2];
    const float* id_conv_w = (const float*)d_in[3];
    const float* id_conv_b = (const float*)d_in[4];
    const float* id_xproj  = (const float*)d_in[5];
    const float* id_dt_w   = (const float*)d_in[6];
    const float* id_dt_b   = (const float*)d_in[7];
    const float* id_A_log  = (const float*)d_in[8];
    const float* id_Dv     = (const float*)d_in[9];
    const float* id_out_w  = (const float*)d_in[10];
    const float* at_in_w   = (const float*)d_in[11];
    const float* at_conv_w = (const float*)d_in[12];
    const float* at_conv_b = (const float*)d_in[13];
    const float* at_xproj  = (const float*)d_in[14];
    const float* at_dt_w   = (const float*)d_in[15];
    const float* at_dt_b   = (const float*)d_in[16];
    const float* at_A_log  = (const float*)d_in[17];
    const float* at_Dv     = (const float*)d_in[18];
    const float* at_out_w  = (const float*)d_in[19];
    const float* id_ln_g   = (const float*)d_in[20];
    const float* id_ln_b   = (const float*)d_in[21];
    const float* at_ln_g   = (const float*)d_in[22];
    const float* at_ln_b   = (const float*)d_in[23];
    float* outp = (float*)d_out;

    char* ws = (char*)d_ws;
    size_t off = 0;
    auto alloc = [&](size_t elems, size_t esz) -> void* {
        void* p = ws + off;
        off += ((elems * esz + 255) & ~(size_t)255);
        return p;
    };

    const int L = L_SEQ, D = DMODEL;

    // ---- id stream buffers ----
    bf16* iw_in  = (bf16*)alloc((size_t)2 * DI_ID * D, 2);
    bf16* iw_xp  = (bf16*)alloc((size_t)R_ID * DI_ID, 2);
    bf16* iw_dt  = (bf16*)alloc((size_t)DI_ID * DTR, 2);
    bf16* iw_out = (bf16*)alloc((size_t)D * DI_ID, 2);
    bf16* i_xn   = (bf16*)alloc((size_t)L * D, 2);
    float* i_xz  = (float*)alloc((size_t)L * 2 * DI_ID, 4);
    float* i_xif = (float*)alloc((size_t)L * DI_ID, 4);
    bf16* i_xib  = (bf16*)alloc((size_t)L * DI_ID, 2);
    float* i_dbl = (float*)alloc((size_t)L * R_ID, 4);
    bf16* i_dtb  = (bf16*)alloc((size_t)L * DTR, 2);
    float* i_del = (float*)alloc((size_t)L * DI_ID, 4);
    float* i_ys  = (float*)alloc((size_t)L * DI_ID, 4);
    bf16* i_yb   = (bf16*)alloc((size_t)L * DI_ID, 2);
    float* i_out = (float*)alloc((size_t)L * D, 4);

    // ---- attr stream buffers ----
    bf16* aw_in  = (bf16*)alloc((size_t)2 * DI_AT * D, 2);
    bf16* aw_xp  = (bf16*)alloc((size_t)R_AT * DI_AT, 2);
    bf16* aw_dt  = (bf16*)alloc((size_t)DI_AT * DTR, 2);
    bf16* aw_out = (bf16*)alloc((size_t)D * DI_AT, 2);
    bf16* a_xn   = (bf16*)alloc((size_t)L * D, 2);
    float* a_xz  = (float*)alloc((size_t)L * 2 * DI_AT, 4);
    float* a_xif = (float*)alloc((size_t)L * DI_AT, 4);
    bf16* a_xib  = (bf16*)alloc((size_t)L * DI_AT, 2);
    float* a_dbl = (float*)alloc((size_t)L * R_AT, 4);
    bf16* a_dtb  = (bf16*)alloc((size_t)L * DTR, 2);
    float* a_del = (float*)alloc((size_t)L * DI_AT, 4);
    float* a_ys  = (float*)alloc((size_t)L * DI_AT, 4);
    bf16* a_yb   = (bf16*)alloc((size_t)L * DI_AT, 2);
    float* a_out = (float*)alloc((size_t)L * D, 4);

    run_stream(stream, id_seq, id_ln_g, id_ln_b,
               id_in_w, id_conv_w, id_conv_b, id_xproj, id_dt_w, id_dt_b,
               id_A_log, id_Dv, id_out_w,
               DI_ID, NST_ID, R_ID,
               iw_in, iw_xp, iw_dt, iw_out,
               i_xn, i_xz, i_xif, i_xib, i_dbl, i_dtb, i_del, i_ys, i_yb,
               i_out, outp);

    run_stream(stream, at_seq, at_ln_g, at_ln_b,
               at_in_w, at_conv_w, at_conv_b, at_xproj, at_dt_w, at_dt_b,
               at_A_log, at_Dv, at_out_w,
               DI_AT, NST_AT, R_AT,
               aw_in, aw_xp, aw_dt, aw_out,
               a_xn, a_xz, a_xif, a_xib, a_dbl, a_dtb, a_del, a_ys, a_yb,
               a_out, outp + (size_t)L * D);
}
